// RV2BEVFrustumAttn_55095840473532
// MI455X (gfx1250) — compile-verified, matrix operand order checked by reading, not run on hardware
//
#include <hip/hip_runtime.h>

// ---------------- problem constants ----------------
#define B_     2
#define HRV    16
#define WRV    1024
#define CRV    64
#define POSB   16384          // Hrv*Wrv
#define NPOS   32768          // B*Hrv*Wrv
#define HBEV   220
#define WBEV   220
#define HWBEV  48400
#define NBEV   96800          // B*Hbev*Wbev
#define DMODEL 128
#define KLVL   5
#define NH_    8
#define NP_    4
#define DH_    16
#define PI_F   3.14159265358979323846f
#define AZSTEP 0.006135923151542565f   // 2*pi/1024
#define D2R    0.017453292519943295f
#define RMAXF  55.0f

typedef __bf16 bf16;
typedef bf16  v16bf __attribute__((ext_vector_type(16)));
typedef bf16  v8bf  __attribute__((ext_vector_type(8)));
typedef float v8f   __attribute__((ext_vector_type(8)));

union frag_u { v16bf v; v8bf h[2]; };

__device__ __constant__ float c_elev[32] = {
  -30.67f,-29.33f,-28.0f,-26.66f,-25.33f,-24.0f,-22.67f,-21.33f,-20.0f,-18.67f,
  -17.33f,-16.0f,-14.67f,-13.33f,-12.0f,-10.67f,-9.33f,-8.0f,-6.66f,-5.33f,
  -4.0f,-2.67f,-1.33f,0.0f,1.33f,2.67f,4.0f,5.33f,6.67f,8.0f,9.33f,10.67f };

__device__ __forceinline__ float elev_mean_of(int h) {
  return D2R * 0.5f * (c_elev[31-2*h] + c_elev[30-2*h]);
}
__device__ __forceinline__ float elev_step_of(int i) {   // i = w % 16
  return D2R * (c_elev[31-2*i] - c_elev[30-2*i]);
}
__device__ __forceinline__ float az_of(int w) { return -PI_F + AZSTEP*((float)w + 0.5f); }

__device__ __forceinline__ float gelu_exact(float x) {
  return 0.5f * x * (1.0f + erff(x * 0.7071067811865476f));
}
__device__ __forceinline__ float sigmf(float x)  { return 1.0f / (1.0f + expf(-x)); }
__device__ __forceinline__ float softplusf(float x) { return (x > 20.f) ? x : log1pf(expf(x)); }

// Load one 16-element bf16 fragment: two contiguous 8-elem (16B) chunks.
// Per ISA layout, lane-half h needs elements [k0+8h, +8) and [k0+16+8h, +8).
__device__ __forceinline__ v16bf load_frag(const bf16* __restrict__ row, int k0, int half) {
  frag_u u;
  u.h[0] = *reinterpret_cast<const v8bf*>(row + k0 + 8*half);
  u.h[1] = *reinterpret_cast<const v8bf*>(row + k0 + 16 + 8*half);
  return u.v;
}

// ====================================================================
// bf16 WMMA GEMM, fully unrolled over K (CINX = padded Cin, mult of 32).
//   X: bf16 row-major [Nrows x ldx]; Wt: bf16 row-major [Cout x ldw].
//   Yf (optional): fp32 with (pos,chan,batch) strides (NCHW-flat capable)
//   Yb (optional): bf16 row-major [Nrows x Cout]
//   act: 0=none, 1=gelu ; resid: optional fp32 row-major [Nrows x Cout]
//   Requires Nrows % 16 == 0 and Cout % 16 == 0 (true for all call sites).
// ====================================================================
template<int CINX>
__global__ void gemm_wmma(const bf16* __restrict__ X, int ldx,
                          const bf16* __restrict__ Wt, int ldw,
                          const float* __restrict__ bias,
                          const float* __restrict__ resid,
                          float* __restrict__ Yf, long long yPos, long long yChan,
                          long long yBatch, int rowsPerBatchOut,
                          bf16* __restrict__ Yb,
                          int Nrows, int Cout, int act)
{
  int wid  = (int)((blockIdx.x * blockDim.x + threadIdx.x) >> 5);
  int lane = threadIdx.x & 31;
  int nRowT = Nrows >> 4;
  int nColT = Cout >> 4;
  if (wid >= nRowT * nColT) return;
  int tr = wid % nRowT, tc = wid / nRowT;
  int half = lane >> 4, l16 = lane & 15;

  const bf16* xp = X  + (long long)(tr*16 + l16) * ldx;
  const bf16* wp = Wt + (long long)(tc*16 + l16) * ldw;

  v8f acc = {};
#pragma unroll
  for (int k0 = 0; k0 < CINX; k0 += 32) {
    v16bf afr = load_frag(xp, k0, half);
    v16bf bfr = load_frag(wp, k0, half);
    acc = __builtin_amdgcn_wmma_f32_16x16x32_bf16(false, afr, false, bfr,
                                                  (short)0, acc, false, false);
  }

  int ocol = tc*16 + l16;
  float bv = bias ? bias[ocol] : 0.f;
#pragma unroll
  for (int v = 0; v < 8; ++v) {
    int r = tr*16 + v + 8*half;
    float val = acc[v] + bv;
    if (resid) val += resid[(long long)r * Cout + ocol];
    if (act == 1) val = gelu_exact(val);
    if (Yf) {
      int b = r / rowsPerBatchOut, p = r - b*rowsPerBatchOut;
      Yf[(long long)b*yBatch + (long long)p*yPos + (long long)ocol*yChan] = val;
    }
    if (Yb) Yb[(long long)r * Cout + ocol] = (bf16)val;
  }
}

// ====================================================================
// 3x3 conv, circular pad (H,W), 128 -> 64 ch, no bias, WMMA.
//   Xin bf16 row-major [NPOS x 128]
//   Wc  bf16 tap-major [9][64][128]  (pre-packed)
//   Y   fp32 [NPOS x 64]
// ====================================================================
__global__ void conv3x3_wmma(const bf16* __restrict__ Xin,
                             const bf16* __restrict__ Wc,
                             float* __restrict__ Y, int Nrows)
{
  const int Cin = 128, Cout = 64;
  int wid  = (int)((blockIdx.x * blockDim.x + threadIdx.x) >> 5);
  int lane = threadIdx.x & 31;
  int nRowT = Nrows >> 4, nColT = Cout >> 4;
  if (wid >= nRowT * nColT) return;
  int tr = wid % nRowT, tc = wid / nRowT;
  int half = lane >> 4, l16 = lane & 15;

  int arow = tr*16 + l16;
  int b  = arow >> 14;
  int rm = arow & (POSB - 1);
  int hh = rm >> 10, ww = rm & (WRV - 1);
  int ocol = tc*16 + l16;

  v8f acc = {};
#pragma unroll
  for (int t = 0; t < 9; ++t) {
    int dy = t/3 - 1, dx = t%3 - 1;
    int nh = (hh + dy + HRV) & (HRV - 1);
    int nw = (ww + dx + WRV) & (WRV - 1);
    const bf16* xp = Xin + (long long)((b << 14) + (nh << 10) + nw) * Cin;
    const bf16* wp = Wc  + (long long)(t*Cout + ocol) * Cin;
#pragma unroll
    for (int k0 = 0; k0 < Cin; k0 += 32) {
      v16bf afr = load_frag(xp, k0, half);
      v16bf bfr = load_frag(wp, k0, half);
      acc = __builtin_amdgcn_wmma_f32_16x16x32_bf16(false, afr, false, bfr,
                                                    (short)0, acc, false, false);
    }
  }
#pragma unroll
  for (int v = 0; v < 8; ++v) {
    int r = tr*16 + v + 8*half;
    Y[(long long)r * Cout + ocol] = acc[v];
  }
}

// ====================================================================
// packing / conversion kernels
// ====================================================================

// fp32 [Cout x CinW] -> bf16 zero-padded [Cout x CinX]
__global__ void pack_w(const float* __restrict__ W, bf16* __restrict__ Wb,
                       int CinW, int CinX, long long total)
{
  long long i = (long long)blockIdx.x * blockDim.x + threadIdx.x;
  if (i >= total) return;
  int k = (int)(i % CinX);
  long long o = i / CinX;
  Wb[i] = (k < CinW) ? (bf16)W[o*CinW + k] : (bf16)0.0f;
}

// rh_w2 [64][128][3][3] -> bf16 tap-major [9][64][128]
__global__ void pack_convw(const float* __restrict__ W, bf16* __restrict__ Wb)
{
  int i = blockIdx.x * blockDim.x + threadIdx.x;   // 9*64*128
  if (i >= 9*64*128) return;
  int c = i & 127, o = (i >> 7) & 63, t = i >> 13;
  Wb[i] = (bf16)W[o*1152 + c*9 + t];
}

// bev NCHW fp32 -> row-major bf16 [NBEV x 128]
__global__ void pack_bev(const float* __restrict__ bev, bf16* __restrict__ out)
{
  long long i = (long long)blockIdx.x * blockDim.x + threadIdx.x;
  if (i >= (long long)NBEV * 128) return;
  int c = (int)(i & 127);
  long long n = i >> 7;
  int b = (int)(n / HWBEV), hw = (int)(n - (long long)b*HWBEV);
  out[i] = (bf16)bev[(long long)b*128*HWBEV + (long long)c*HWBEV + hw];
}

// range-head input: bf16 [NPOS x 96] = [x(64) | u_vec(3) | zeros]
__global__ void pack_rin(const float* __restrict__ x_rv, bf16* __restrict__ rin)
{
  long long i = (long long)blockIdx.x * blockDim.x + threadIdx.x;
  if (i >= (long long)NPOS * 96) return;
  long long n = i / 96; int j = (int)(i % 96);
  float v = 0.f;
  if (j < CRV) v = x_rv[n*CRV + j];
  else if (j < 67) {
    int rm = (int)(n & (POSB - 1));
    int h = rm >> 10, w = rm & (WRV - 1);
    float el = elev_mean_of(h), az = az_of(w);
    float ce = cosf(el);
    v = (j == 64) ? ce*cosf(az) : (j == 65) ? ce*sinf(az) : sinf(el);
  }
  rin[i] = (bf16)v;
}

// qs-MLP input: bf16 [NPOS x 160] = [Q0(128) | sig(6) | zeros]
__global__ void pack_qcat(const bf16* __restrict__ Q0b, const float* __restrict__ sig,
                          bf16* __restrict__ qcat)
{
  long long i = (long long)blockIdx.x * blockDim.x + threadIdx.x;
  if (i >= (long long)NPOS * 160) return;
  long long n = i / 160; int j = (int)(i % 160);
  bf16 v = (bf16)0.0f;
  if (j < DMODEL) v = Q0b[n*DMODEL + j];
  else if (j < DMODEL + 6) v = (bf16)sig[n*6 + (j - DMODEL)];
  qcat[i] = v;
}

// fold w_v into val_w -> bf16 Wvv [128 x 128], fp32 bvv
__global__ void fuse_vw(const float* __restrict__ val_w, const float* __restrict__ w_v,
                        const float* __restrict__ val_b, const float* __restrict__ b_v,
                        bf16* __restrict__ Wvv, float* __restrict__ bvv)
{
  int i = blockIdx.x * blockDim.x + threadIdx.x;
  if (i >= DMODEL * DMODEL) return;
  int o = i >> 7, cb = i & 127;
  float s = 0.f;
  for (int c = 0; c < DMODEL; ++c) s += val_w[o*DMODEL + c] * w_v[c*DMODEL + cb];
  Wvv[i] = (bf16)s;
  if (cb == 0) {
    float t = val_b[o];
    for (int c = 0; c < DMODEL; ++c) t += val_w[o*DMODEL + c] * b_v[c];
    bvv[o] = t;
  }
}

// ====================================================================
// GroupNorm stats: X fp32 row-major [B*posPerBatch x C]; one block per (b,g)
// ====================================================================
__global__ void gn_reduce(const float* __restrict__ X, int C, int grpCh,
                          int posPerBatch, float* __restrict__ stats)
{
  int bg = blockIdx.x;
  int G = C / grpCh;
  int b = bg / G, g = bg % G;
  long long total = (long long)posPerBatch * grpCh;
  float s = 0.f, s2 = 0.f;
  for (long long i = threadIdx.x; i < total; i += blockDim.x) {
    int p = (int)(i / grpCh), c = g*grpCh + (int)(i % grpCh);
    float v = X[((long long)b*posPerBatch + p)*C + c];
    s += v; s2 += v*v;
  }
  __shared__ float sh[512];
  sh[threadIdx.x] = s; sh[threadIdx.x + 256] = s2;
  __syncthreads();
  for (int o = 128; o > 0; o >>= 1) {
    if ((int)threadIdx.x < o) { sh[threadIdx.x] += sh[threadIdx.x+o]; sh[threadIdx.x+256] += sh[threadIdx.x+256+o]; }
    __syncthreads();
  }
  if (threadIdx.x == 0) {
    float mean = sh[0] / (float)total;
    float var  = sh[256] / (float)total - mean*mean;
    stats[2*bg]   = mean;
    stats[2*bg+1] = rsqrtf(var + 1e-5f);
  }
}

// GN + gelu; writes fp32 (optional) and bf16 (optional) copies
__global__ void gn_apply_gelu(const float* __restrict__ X,
                              const float* __restrict__ gamma, const float* __restrict__ beta,
                              const float* __restrict__ stats, int C, int grpCh,
                              int posPerBatch, long long total,
                              float* __restrict__ outf, bf16* __restrict__ outb)
{
  long long i = (long long)blockIdx.x * blockDim.x + threadIdx.x;
  if (i >= total) return;
  int c = (int)(i % C);
  long long n = i / C;
  int b = (int)(n / posPerBatch);
  int G = C / grpCh;
  int bg = b*G + c / grpCh;
  float v = (X[i] - stats[2*bg]) * stats[2*bg+1] * gamma[c] + beta[c];
  v = gelu_exact(v);
  if (outf) outf[i] = v;
  if (outb) outb[i] = (bf16)v;
}

// ---------------- range head 1x1 (64->4) + geometry + ref points + KL
__global__ void head_geom(const float* __restrict__ h2,
                          const float* __restrict__ rh_w3, const float* __restrict__ rh_b3,
                          const float* __restrict__ l2e,
                          float* __restrict__ sig, float* __restrict__ ref,
                          float* __restrict__ klacc)
{
  int n = blockIdx.x * blockDim.x + threadIdx.x;
  float kl_local = 0.f;
  if (n < NPOS) {
    const float* hp = h2 + (long long)n * 64;
    float hd[4];
#pragma unroll
    for (int j = 0; j < 4; ++j) {
      float a = rh_b3[j];
      for (int c = 0; c < 64; ++c) a += rh_w3[j*64 + c] * hp[c];
      hd[j] = a;
    }
    int rm = n & (POSB - 1);
    int hI = rm >> 10, wI = rm & (WRV - 1);
    float estep = elev_step_of(wI & 15);
    float emean = elev_mean_of(hI);
    float az0   = az_of(wI);

    float mu = sigmf(hd[0]) * RMAXF;
    float sd = softplusf(hd[1]) + 1e-3f;
    float se = sigmf(hd[2]) * estep + 1e-3f;
    float sa = sigmf(hd[3]) * AZSTEP + 1e-3f;

    sig[n*6+0] = sd; sig[n*6+1] = se; sig[n*6+2] = sa;
    sig[n*6+3] = 1.f/(sd+1e-6f); sig[n*6+4] = 1.f/(se+1e-6f); sig[n*6+5] = 1.f/(sa+1e-6f);

#pragma unroll
    for (int k = 0; k < KLVL; ++k) {
      float off = -0.6f + 0.3f * (float)k;
      float depth = mu + sd*off;
      float el    = emean + se*off;
      float az    = az0 + sa*off;
      float ce = cosf(el);
      float xl = depth*ce*cosf(az), yl = depth*ce*sinf(az), zl = depth*sinf(el);
      float px = xl*l2e[0] + yl*l2e[4] + zl*l2e[8]  + l2e[12];
      float py = xl*l2e[1] + yl*l2e[5] + zl*l2e[9]  + l2e[13];
      float rx = sigmf(4.0f * ((px + 55.f) / 110.f - 0.5f));
      float ry = sigmf(4.0f * ((py + 55.f) / 110.f - 0.5f));
      ref[(long long)n*(KLVL*2) + k*2 + 0] = rx;
      ref[(long long)n*(KLVL*2) + k*2 + 1] = ry;
    }
    float mun = mu / RMAXF;
    float sr  = fmaxf(sd, 1e-6f);
    float saz = sa / AZSTEP, sel = se / estep;
    kl_local = 0.5f*(mun*mun + sr*sr - logf(sr*sr) - 1.f)
             + 0.5f*(saz*saz - logf(saz*saz) - 1.f)
             + 0.5f*(sel*sel - logf(sel*sel) - 1.f);
  }
  __shared__ float red[256];
  red[threadIdx.x] = kl_local;
  __syncthreads();
  for (int o = 128; o > 0; o >>= 1) {
    if ((int)threadIdx.x < o) red[threadIdx.x] += red[threadIdx.x + o];
    __syncthreads();
  }
  if (threadIdx.x == 0) atomicAdd(klacc, red[0]);
}

// ---------------- softmax over K*NP=20 per (n, head), in place
__global__ void softmax20(float* __restrict__ attn)
{
  int t = blockIdx.x * blockDim.x + threadIdx.x;
  if (t >= NPOS * NH_) return;
  int n = t >> 3, h = t & 7;
  float* p = attn + (long long)n * (NH_*KLVL*NP_) + h * (KLVL*NP_);
  float mx = -1e30f;
#pragma unroll
  for (int i = 0; i < 20; ++i) mx = fmaxf(mx, p[i]);
  float s = 0.f;
  float e[20];
#pragma unroll
  for (int i = 0; i < 20; ++i) { e[i] = expf(p[i] - mx); s += e[i]; }
  float inv = 1.f / s;
#pragma unroll
  for (int i = 0; i < 20; ++i) p[i] = e[i] * inv;
}

// ---------------- MSDA bilinear gather; one thread per (n, head); bf16 out
__global__ void msda_sample(const float* __restrict__ v,
                            const float* __restrict__ ref,
                            const float* __restrict__ smp,
                            const float* __restrict__ attn,
                            bf16* __restrict__ outp)
{
  int t = blockIdx.x * blockDim.x + threadIdx.x;
  if (t >= NPOS * NH_) return;
  int n = t >> 3, h = t & 7;
  int b = n >> 14;
  const float* vb = v + (long long)b * HWBEV * DMODEL + h * DH_;

  float acc[16];
#pragma unroll
  for (int d = 0; d < 16; ++d) acc[d] = 0.f;

  for (int k = 0; k < KLVL; ++k) {
    float rx = ref[(long long)n*(KLVL*2) + k*2 + 0];
    float ry = ref[(long long)n*(KLVL*2) + k*2 + 1];
#pragma unroll
    for (int p = 0; p < NP_; ++p) {
      long long oidx = (long long)n*320 + (long long)(((h*KLVL + k)*NP_ + p)*2);
      float lx = rx + smp[oidx]     * (1.0f / WBEV);
      float ly = ry + smp[oidx + 1] * (1.0f / HBEV);
      float w  = attn[(long long)n*160 + h*20 + k*4 + p];
      float x = lx * WBEV - 0.5f, y = ly * HBEV - 0.5f;
      float xf = floorf(x), yf = floorf(y);
      float wx = x - xf, wy = y - yf;
      int x0 = (int)xf, y0 = (int)yf;
#pragma unroll
      for (int cy = 0; cy < 2; ++cy) {
#pragma unroll
        for (int cx = 0; cx < 2; ++cx) {
          int xi = x0 + cx, yi = y0 + cy;
          if (xi >= 0 && xi < WBEV && yi >= 0 && yi < HBEV) {
            float cw = w * (cx ? wx : 1.f - wx) * (cy ? wy : 1.f - wy);
            const float4* vp4 = reinterpret_cast<const float4*>(vb + (long long)(yi*WBEV + xi) * DMODEL);
#pragma unroll
            for (int q = 0; q < 4; ++q) {
              float4 f = vp4[q];
              acc[4*q+0] += cw * f.x; acc[4*q+1] += cw * f.y;
              acc[4*q+2] += cw * f.z; acc[4*q+3] += cw * f.w;
            }
          }
        }
      }
    }
  }
  bf16* op = outp + (long long)n * DMODEL + h * DH_;
#pragma unroll
  for (int d = 0; d < 16; ++d) op[d] = (bf16)acc[d];
}

__global__ void zero_kl(float* klacc) { *klacc = 0.f; }
__global__ void write_kl(const float* klacc, float* d_out) {
  d_out[(long long)NPOS * DMODEL] = 1e-4f * (*klacc) * (1.0f / (float)NPOS);
}

// ====================================================================
extern "C" void kernel_launch(void* const* d_in, const int* in_sizes, int n_in,
                              void* d_out, int out_size, void* d_ws, size_t ws_size,
                              hipStream_t stream)
{
  const float* x_rv  = (const float*)d_in[0];
  const float* bev   = (const float*)d_in[1];
  const float* l2e   = (const float*)d_in[2];
  const float* w_q   = (const float*)d_in[3];
  const float* b_q   = (const float*)d_in[4];
  const float* w_v   = (const float*)d_in[5];
  const float* b_v   = (const float*)d_in[6];
  const float* w_o   = (const float*)d_in[7];
  const float* b_o   = (const float*)d_in[8];
  const float* qs_w1 = (const float*)d_in[9];
  const float* qs_b1 = (const float*)d_in[10];
  const float* qs_w2 = (const float*)d_in[11];
  const float* qs_b2 = (const float*)d_in[12];
  const float* rh_w1 = (const float*)d_in[13];
  const float* rh_b1 = (const float*)d_in[14];
  const float* rh_g1 = (const float*)d_in[15];
  const float* rh_bt1= (const float*)d_in[16];
  const float* rh_w2 = (const float*)d_in[17];
  const float* rh_g2 = (const float*)d_in[18];
  const float* rh_bt2= (const float*)d_in[19];
  const float* rh_w3 = (const float*)d_in[20];
  const float* rh_b3 = (const float*)d_in[21];
  const float* off_w = (const float*)d_in[22];
  const float* off_b = (const float*)d_in[23];
  const float* aw_w  = (const float*)d_in[24];
  const float* aw_b  = (const float*)d_in[25];
  const float* val_w = (const float*)d_in[26];
  const float* val_b = (const float*)d_in[27];
  const float* out_w = (const float*)d_in[28];
  const float* out_b = (const float*)d_in[29];
  float* out = (float*)d_out;
  (void)ws_size; (void)n_in; (void)in_sizes; (void)out_size;

  // ---- workspace carve (bytes, 256B aligned per buffer) ----
  char* base = (char*)d_ws;
  size_t off = 0;
  auto alloc = [&](size_t bytes) {
    off = (off + 255) & ~(size_t)255;
    void* p = base + off;
    off += bytes;
    return p;
  };
  // bf16 activation buffers
  bf16* xb     = (bf16*)alloc((size_t)NPOS * 64  * 2);
  bf16* rinb   = (bf16*)alloc((size_t)NPOS * 96  * 2);
  bf16* Q0b    = (bf16*)alloc((size_t)NPOS * 128 * 2);
  bf16* h1b    = (bf16*)alloc((size_t)NPOS * 128 * 2);
  bf16* qcatb  = (bf16*)alloc((size_t)NPOS * 160 * 2);
  bf16* qhb    = (bf16*)alloc((size_t)NPOS * 128 * 2);
  bf16* queryb = (bf16*)alloc((size_t)NPOS * 128 * 2);
  bf16* bevb   = (bf16*)alloc((size_t)NBEV * 128 * 2);
  bf16* msdab  = (bf16*)alloc((size_t)NPOS * 128 * 2);
  bf16* ymsdab = (bf16*)alloc((size_t)NPOS * 128 * 2);
  // bf16 packed weights
  bf16* wq_b   = (bf16*)alloc((size_t)128 * 64  * 2);
  bf16* rhw1_b = (bf16*)alloc((size_t)128 * 96  * 2);
  bf16* qsw1_b = (bf16*)alloc((size_t)128 * 160 * 2);
  bf16* qsw2_b = (bf16*)alloc((size_t)128 * 128 * 2);
  bf16* offw_b = (bf16*)alloc((size_t)320 * 128 * 2);
  bf16* aww_b  = (bf16*)alloc((size_t)160 * 128 * 2);
  bf16* outw_b = (bf16*)alloc((size_t)128 * 128 * 2);
  bf16* wo_b   = (bf16*)alloc((size_t)128 * 128 * 2);
  bf16* convw_b= (bf16*)alloc((size_t)9*64*128 * 2);
  bf16* wvv_b  = (bf16*)alloc((size_t)128 * 128 * 2);
  // fp32 buffers
  float* h1    = (float*)alloc((size_t)NPOS * 128 * 4);
  float* h2    = (float*)alloc((size_t)NPOS * 64  * 4);
  float* sig   = (float*)alloc((size_t)NPOS * 6   * 4);
  float* ref   = (float*)alloc((size_t)NPOS * 10  * 4);
  float* query = (float*)alloc((size_t)NPOS * 128 * 4);
  float* vbuf  = (float*)alloc((size_t)NBEV * 128 * 4);
  float* smp   = (float*)alloc((size_t)NPOS * 320 * 4);
  float* attn  = (float*)alloc((size_t)NPOS * 160 * 4);
  float* bvv   = (float*)alloc(128 * 4);
  float* stats = (float*)alloc(64 * 4);
  float* klacc = (float*)alloc(4);

  const int TB = 256;
  auto cdiv = [](long long a, long long b) { return (int)((a + b - 1) / b); };
  auto gemmBlocks = [](int Nrows, int Cout) {
    long long waves = (long long)(Nrows / 16) * (Cout / 16);
    return (int)((waves + 7) / 8);
  };

  zero_kl<<<1, 1, 0, stream>>>(klacc);

  // ---- one-shot packing / conversion ----
  fuse_vw<<<cdiv(128*128, TB), TB, 0, stream>>>(val_w, w_v, val_b, b_v, wvv_b, bvv);
  pack_w<<<cdiv(128*64 , TB), TB, 0, stream>>>(w_q  , wq_b  , 64 , 64 , 128*64 );
  pack_w<<<cdiv(128*96 , TB), TB, 0, stream>>>(rh_w1, rhw1_b, 67 , 96 , 128*96 );
  pack_w<<<cdiv(128*160, TB), TB, 0, stream>>>(qs_w1, qsw1_b, 134, 160, 128*160);
  pack_w<<<cdiv(128*128, TB), TB, 0, stream>>>(qs_w2, qsw2_b, 128, 128, 128*128);
  pack_w<<<cdiv(320*128, TB), TB, 0, stream>>>(off_w, offw_b, 128, 128, (long long)320*128);
  pack_w<<<cdiv(160*128, TB), TB, 0, stream>>>(aw_w , aww_b , 128, 128, (long long)160*128);
  pack_w<<<cdiv(128*128, TB), TB, 0, stream>>>(out_w, outw_b, 128, 128, 128*128);
  pack_w<<<cdiv(128*128, TB), TB, 0, stream>>>(w_o  , wo_b  , 128, 128, 128*128);
  pack_convw<<<cdiv(9*64*128, TB), TB, 0, stream>>>(rh_w2, convw_b);
  pack_w<<<cdiv((long long)NPOS*64, TB), TB, 0, stream>>>(x_rv, xb, 64, 64, (long long)NPOS*64);
  pack_bev<<<cdiv((long long)NBEV*128, TB), TB, 0, stream>>>(bev, bevb);
  pack_rin<<<cdiv((long long)NPOS*96, TB), TB, 0, stream>>>(x_rv, rinb);

  // ---- Q0 = x @ w_q^T + b_q  (bf16 out) ----
  gemm_wmma<64><<<gemmBlocks(NPOS,128), TB, 0, stream>>>(
      xb, 64, wq_b, 64, b_q, nullptr,
      nullptr, 0, 0, 0, NPOS, Q0b, NPOS, 128, 0);

  // ---- range head layer 1 (fp32 out for GN) ----
  gemm_wmma<96><<<gemmBlocks(NPOS,128), TB, 0, stream>>>(
      rinb, 96, rhw1_b, 96, rh_b1, nullptr,
      h1, 128, 1, 0, NPOS, nullptr, NPOS, 128, 0);
  gn_reduce<<<16, TB, 0, stream>>>(h1, 128, 16, POSB, stats);
  gn_apply_gelu<<<cdiv((long long)NPOS*128, TB), TB, 0, stream>>>(
      h1, rh_g1, rh_bt1, stats, 128, 16, POSB, (long long)NPOS*128, nullptr, h1b);

  // ---- 3x3 circular conv (WMMA) -> fp32 h2 ----
  conv3x3_wmma<<<gemmBlocks(NPOS,64), TB, 0, stream>>>(h1b, convw_b, h2, NPOS);
  gn_reduce<<<16, TB, 0, stream>>>(h2, 64, 8, POSB, stats + 32);
  gn_apply_gelu<<<cdiv((long long)NPOS*64, TB), TB, 0, stream>>>(
      h2, rh_g2, rh_bt2, stats + 32, 64, 8, POSB, (long long)NPOS*64, h2, nullptr);

  // ---- head + geometry + ref + KL ----
  head_geom<<<NPOS/TB, TB, 0, stream>>>(h2, rh_w3, rh_b3, l2e, sig, ref, klacc);

  // ---- qs MLP ----
  pack_qcat<<<cdiv((long long)NPOS*160, TB), TB, 0, stream>>>(Q0b, sig, qcatb);
  gemm_wmma<160><<<gemmBlocks(NPOS,128), TB, 0, stream>>>(
      qcatb, 160, qsw1_b, 160, qs_b1, nullptr,
      nullptr, 0, 0, 0, NPOS, qhb, NPOS, 128, 1);
  gemm_wmma<128><<<gemmBlocks(NPOS,128), TB, 0, stream>>>(
      qhb, 128, qsw2_b, 128, qs_b2, nullptr,
      query, 128, 1, 0, NPOS, queryb, NPOS, 128, 0);

  // ---- value map (fused w_v ∘ val_w) ----
  gemm_wmma<128><<<gemmBlocks(NBEV,128), TB, 0, stream>>>(
      bevb, 128, wvv_b, 128, bvv, nullptr,
      vbuf, 128, 1, 0, NBEV, nullptr, NBEV, 128, 0);

  // ---- sampling offsets + attention logits ----
  gemm_wmma<128><<<gemmBlocks(NPOS,320), TB, 0, stream>>>(
      queryb, 128, offw_b, 128, off_b, nullptr,
      smp, 320, 1, 0, NPOS, nullptr, NPOS, 320, 0);
  gemm_wmma<128><<<gemmBlocks(NPOS,160), TB, 0, stream>>>(
      queryb, 128, aww_b, 128, aw_b, nullptr,
      attn, 160, 1, 0, NPOS, nullptr, NPOS, 160, 0);
  softmax20<<<(NPOS*NH_)/TB, TB, 0, stream>>>(attn);

  // ---- deformable bilinear gather ----
  msda_sample<<<(NPOS*NH_)/TB, TB, 0, stream>>>(vbuf, ref, smp, attn, msdab);

  // ---- y_msda = msda @ out_w^T + out_b + query (bf16 out) ----
  gemm_wmma<128><<<gemmBlocks(NPOS,128), TB, 0, stream>>>(
      msdab, 128, outw_b, 128, out_b, query,
      nullptr, 0, 0, 0, NPOS, ymsdab, NPOS, 128, 2);

  // ---- y = ymsda @ w_o^T + b_o, NCHW-flat into d_out ----
  gemm_wmma<128><<<gemmBlocks(NPOS,128), TB, 0, stream>>>(
      ymsdab, 128, wo_b, 128, b_o, nullptr,
      out, 1, POSB, (long long)128*POSB, POSB, nullptr, NPOS, 128, 0);

  write_kl<<<1, 1, 0, stream>>>(klacc, out);
}
